// MoE_9268539425527
// MI455X (gfx1250) — matrix operand order repaired and checked by reading, not compiled, not run on hardware
//
#include <hip/hip_runtime.h>
#include <hip/hip_bf16.h>
#include <stdint.h>

// Problem constants (match reference)
#define BQ    2
#define TQ    2048
#define CDIM  1024
#define EDIM  8
#define FDIM  4096
#define NTOK  (BQ*TQ)       // 4096 tokens
#define NSLOT (NTOK*2)      // 8192 (token,slot) pairs, k=2
#define MAXR  NTOK          // per-expert row capacity
#define POOLR (NSLOT+128)   // pooled rows + tile padding slack

typedef __attribute__((ext_vector_type(16))) __bf16 v16bf;
typedef __attribute__((ext_vector_type(8)))  float  v8f;

union FragU { v16bf v; uint4 q[2]; };

__device__ __forceinline__ unsigned short f2bf(float f) {
    union { float f; unsigned u; } x; x.f = f;
    unsigned r = x.u + 0x7FFFu + ((x.u >> 16) & 1u);   // round-to-nearest-even
    return (unsigned short)(r >> 16);
}

__device__ __forceinline__ float gelu_tanh(float v) {
    float c = 0.7978845608028654f * (v + 0.044715f * v * v * v);
    return 0.5f * v * (1.0f + tanhf(c));
}

// ---------------- prep: fp32 -> bf16 streaming convert (x, W1, W2) ----------
__global__ __launch_bounds__(256) void k_cvt_bf16(const float* __restrict__ src,
                                                  unsigned short* __restrict__ dst, int n) {
    int i = (blockIdx.x * 256 + threadIdx.x) * 4;
    if (i < n) {
        float4 v = *(const float4*)(src + i);
        ushort4 o; o.x = f2bf(v.x); o.y = f2bf(v.y); o.z = f2bf(v.z); o.w = f2bf(v.w);
        *(ushort4*)(dst + i) = o;
    }
}

// ---------------- prep: fold NS=2 shared slabs: WsSum[c][j] = Ws[c][j]+Ws[c][C+j]
__global__ __launch_bounds__(256) void k_prep_ws(const float* __restrict__ Ws,
                                                 const float* __restrict__ bs,
                                                 unsigned short* __restrict__ WsSum,
                                                 float* __restrict__ bsSum) {
    int i = blockIdx.x * 256 + threadIdx.x;      // over C*C
    int c = i >> 10, j = i & 1023;
    WsSum[i] = f2bf(Ws[c * 2048 + j] + Ws[c * 2048 + 1024 + j]);
    if (i < CDIM) bsSum[i] = bs[i] + bs[CDIM + i];
}

// ---------------- gate: one wave per token, sigmoid scores, top-2 ----------
__global__ __launch_bounds__(256) void k_gate(const float* __restrict__ x,
                                              const float* __restrict__ Wg,
                                              const float* __restrict__ bg,
                                              int* __restrict__ slot_e,
                                              float* __restrict__ slot_g,
                                              float* __restrict__ slot_ssel) {
    int wave = threadIdx.x >> 5, lane = threadIdx.x & 31;
    int t = blockIdx.x * 8 + wave;
    float acc[EDIM];
#pragma unroll
    for (int e = 0; e < EDIM; e++) acc[e] = 0.f;
    const float* xr = x + (size_t)t * CDIM;
    for (int c = lane; c < CDIM; c += 32) {
        float xv = xr[c];
        const float* w = Wg + c * EDIM;
#pragma unroll
        for (int e = 0; e < EDIM; e++) acc[e] += xv * w[e];
    }
#pragma unroll
    for (int e = 0; e < EDIM; e++)
        for (int off = 16; off; off >>= 1) acc[e] += __shfl_down(acc[e], off, 32);
    if (lane == 0) {
        float s[EDIM], sum = 0.f;
#pragma unroll
        for (int e = 0; e < EDIM; e++) { s[e] = 1.f / (1.f + expf(-(acc[e] + bg[e]))); sum += s[e]; }
        int i0 = 0;
#pragma unroll
        for (int e = 1; e < EDIM; e++) if (s[e] > s[i0]) i0 = e;
        int i1 = (i0 == 0) ? 1 : 0;
#pragma unroll
        for (int e = 0; e < EDIM; e++) if (e != i0 && s[e] > s[i1]) i1 = e;
        float gs = s[i0] + s[i1];
        slot_e[2 * t]     = i0; slot_g[2 * t]     = s[i0] / gs; slot_ssel[2 * t]     = s[i0] / sum;
        slot_e[2 * t + 1] = i1; slot_g[2 * t + 1] = s[i1] / gs; slot_ssel[2 * t + 1] = s[i1] / sum;
    }
}

// ---------------- deterministic routing: one block per expert, 2-phase scan --
__global__ __launch_bounds__(256) void k_assign(const int* __restrict__ slot_e,
                                                const float* __restrict__ slot_ssel,
                                                int* __restrict__ slot_pos,
                                                int* __restrict__ row_token,
                                                int* __restrict__ cnt,
                                                int* __restrict__ cntB,
                                                float* __restrict__ psumB) {
    int e = blockIdx.x, tid = threadIdx.x;
    __shared__ int   lc[256], lb0[256];
    __shared__ float lp0[256], lp1[256];
    __shared__ int   pref[257];
    const int CH = NSLOT / 256;   // 32 slots per thread, token-major stable order
    int s0 = tid * CH;
    int c = 0, cb0 = 0; float p0 = 0.f, p1 = 0.f;
    for (int i = 0; i < CH; i++) {
        int s = s0 + i;
        if (slot_e[s] == e) {
            c++;
            if (s < NSLOT / 2) { cb0++; p0 += slot_ssel[s]; } else p1 += slot_ssel[s];
        }
    }
    lc[tid] = c; lb0[tid] = cb0; lp0[tid] = p0; lp1[tid] = p1;
    __syncthreads();
    if (tid == 0) { int run = 0; for (int i = 0; i < 256; i++) { pref[i] = run; run += lc[i]; } pref[256] = run; }
    __syncthreads();
    int pos = pref[tid];
    for (int i = 0; i < CH; i++) {
        int s = s0 + i;
        if (slot_e[s] == e) { slot_pos[s] = pos; row_token[e * MAXR + pos] = s >> 1; pos++; }
    }
    if (tid == 0) {
        cnt[e] = pref[256];
        int tb0 = 0; float tp0 = 0.f, tp1 = 0.f;
        for (int i = 0; i < 256; i++) { tb0 += lb0[i]; tp0 += lp0[i]; tp1 += lp1[i]; }
        cntB[0 * EDIM + e] = tb0;             cntB[1 * EDIM + e] = pref[256] - tb0;
        psumB[0 * EDIM + e] = tp0;            psumB[1 * EDIM + e] = tp1;
    }
}

// ---------------- expert prefix bases + load-balance stats (f, p) -----------
__global__ void k_finalize(const int* __restrict__ cnt, const int* __restrict__ cntB,
                           const float* __restrict__ psumB, int* __restrict__ base,
                           float* __restrict__ out_fp) {
    if (threadIdx.x == 0 && blockIdx.x == 0) {
        int run = 0;
        for (int e = 0; e < EDIM; e++) { base[e] = run; run += cnt[e]; }
        for (int b = 0; b < BQ; b++) {
            float tot = 0.f;
            for (int e = 0; e < EDIM; e++) tot += psumB[b * EDIM + e];
            for (int e = 0; e < EDIM; e++) {
                out_fp[b * EDIM + e]            = (float)(TQ * 2 - cntB[b * EDIM + e]); // f
                out_fp[BQ * EDIM + b * EDIM + e] = tot - psumB[b * EDIM + e];           // p
            }
        }
    }
}

// ---------------- shared-path GEMM: res = x + Xb @ WsSum + bsSum ------------
// 128x128 block tile, 8 waves, each wave 64x32 via 4x2 WMMA 16x16x32 bf16
__global__ __launch_bounds__(256) void k_gemm_shared(const unsigned short* __restrict__ Xb,
                                                     const unsigned short* __restrict__ Wt,
                                                     const float* __restrict__ bsSum,
                                                     const float* __restrict__ x,
                                                     float* __restrict__ res) {
    const int K = CDIM, lda = CDIM, ldb = CDIM;
    int m0 = blockIdx.x * 128, n0 = blockIdx.y * 128;
    int w = threadIdx.x >> 5, lane = threadIdx.x & 31;
    int wm = (w & 1) * 64, wn = (w >> 1) * 32;
    v8f acc[4][2];
#pragma unroll
    for (int i = 0; i < 4; i++)
#pragma unroll
        for (int j = 0; j < 2; j++) acc[i][j] = (v8f){0.f,0.f,0.f,0.f,0.f,0.f,0.f,0.f};

    int aRow = m0 + wm + (lane & 15);
    int aK   = (lane >> 4) * 8;
    for (int k0 = 0; k0 < K; k0 += 32) {
        FragU a[4], b[2];
#pragma unroll
        for (int i = 0; i < 4; i++) {
            const unsigned short* pa = Xb + (size_t)(aRow + i * 16) * lda + k0 + aK;
            a[i].q[0] = *(const uint4*)pa; a[i].q[1] = *(const uint4*)(pa + 16);
        }
#pragma unroll
        for (int j = 0; j < 2; j++) {
            const unsigned short* pb = Wt + (size_t)(k0 + aK) * ldb + n0 + wn + j * 16 + (lane & 15);
            for (int s = 0; s < 16; s++) ((unsigned short*)&b[j])[s] = pb[(size_t)((s & 7) + (s >> 3) * 16) * ldb];
        }
#pragma unroll
        for (int i = 0; i < 4; i++)
#pragma unroll
            for (int j = 0; j < 2; j++)
                acc[i][j] = __builtin_amdgcn_wmma_f32_16x16x32_bf16(
                    false, a[i].v, false, b[j].v, (short)0, acc[i][j], false, false);
    }
    int cs = lane & 15, rs = (lane >> 4) * 8;
#pragma unroll
    for (int i = 0; i < 4; i++)
#pragma unroll
        for (int j = 0; j < 2; j++) {
            int col = n0 + wn + j * 16 + cs;
            float bias = bsSum[col];
#pragma unroll
            for (int v = 0; v < 8; v++) {
                int row = m0 + wm + i * 16 + rs + v;
                size_t o = (size_t)row * CDIM + col;
                res[o] = x[o] + acc[i][j][v] + bias;
            }
        }
}

// ---------------- expert FFN-1: H = gelu(gather(Xb) @ W1[e] + b1[e]) --------
__global__ __launch_bounds__(256) void k_ffn1(const unsigned short* __restrict__ Xb,
                                              const unsigned short* __restrict__ W1b,
                                              const float* __restrict__ b1,
                                              const int* __restrict__ row_token,
                                              const int* __restrict__ cnt,
                                              const int* __restrict__ base,
                                              unsigned short* __restrict__ Hpool) {
    int e = blockIdx.z;
    int M = cnt[e];
    int m0 = blockIdx.x * 128;
    if (m0 >= M) return;                       // uniform block-level exit
    int n0 = blockIdx.y * 128;
    const unsigned short* Bw = W1b + (size_t)e * CDIM * FDIM;   // row-major C x F
    int w = threadIdx.x >> 5, lane = threadIdx.x & 31;
    int wm = (w & 1) * 64, wn = (w >> 1) * 32;
    v8f acc[4][2];
#pragma unroll
    for (int i = 0; i < 4; i++)
#pragma unroll
        for (int j = 0; j < 2; j++) acc[i][j] = (v8f){0.f,0.f,0.f,0.f,0.f,0.f,0.f,0.f};

    int toks[4];
    int r = lane & 15;
#pragma unroll
    for (int i = 0; i < 4; i++) {
        int row = m0 + wm + i * 16 + r;
        row = (row < M) ? row : (M - 1);       // clamp ragged tail (stores are guarded)
        toks[i] = row_token[e * MAXR + row];
    }
    int aK = (lane >> 4) * 8;
    for (int k0 = 0; k0 < CDIM; k0 += 32) {
        FragU a[4], b[2];
#pragma unroll
        for (int i = 0; i < 4; i++) {
            const unsigned short* pa = Xb + (size_t)toks[i] * CDIM + k0 + aK;
            a[i].q[0] = *(const uint4*)pa; a[i].q[1] = *(const uint4*)(pa + 16);
        }
#pragma unroll
        for (int j = 0; j < 2; j++) {
            const unsigned short* pb = Bw + (size_t)(k0 + aK) * FDIM + n0 + wn + j * 16 + (lane & 15);
            for (int s = 0; s < 16; s++) ((unsigned short*)&b[j])[s] = pb[(size_t)((s & 7) + (s >> 3) * 16) * FDIM];
        }
#pragma unroll
        for (int i = 0; i < 4; i++)
#pragma unroll
            for (int j = 0; j < 2; j++)
                acc[i][j] = __builtin_amdgcn_wmma_f32_16x16x32_bf16(
                    false, a[i].v, false, b[j].v, (short)0, acc[i][j], false, false);
    }
    int eb = base[e];
    int cs = lane & 15, rs = (lane >> 4) * 8;
#pragma unroll
    for (int i = 0; i < 4; i++)
#pragma unroll
        for (int j = 0; j < 2; j++) {
            int col = n0 + wn + j * 16 + cs;
            float bias = b1[e * FDIM + col];
#pragma unroll
            for (int v = 0; v < 8; v++) {
                int row = m0 + wm + i * 16 + rs + v;
                if (row < M)
                    Hpool[(size_t)(eb + row) * FDIM + col] = f2bf(gelu_tanh(acc[i][j][v] + bias));
            }
        }
}

// ---------------- expert FFN-2: Y = H @ W2[e] + b2[e] -----------------------
__global__ __launch_bounds__(256) void k_ffn2(const unsigned short* __restrict__ Hpool,
                                              const unsigned short* __restrict__ W2b,
                                              const float* __restrict__ b2,
                                              const int* __restrict__ cnt,
                                              const int* __restrict__ base,
                                              float* __restrict__ Ypool) {
    int e = blockIdx.z;
    int M = cnt[e];
    int m0 = blockIdx.x * 128;
    if (m0 >= M) return;
    int n0 = blockIdx.y * 128;
    const unsigned short* Bw = W2b + (size_t)e * FDIM * CDIM;   // row-major F x C
    int w = threadIdx.x >> 5, lane = threadIdx.x & 31;
    int wm = (w & 1) * 64, wn = (w >> 1) * 32;
    v8f acc[4][2];
#pragma unroll
    for (int i = 0; i < 4; i++)
#pragma unroll
        for (int j = 0; j < 2; j++) acc[i][j] = (v8f){0.f,0.f,0.f,0.f,0.f,0.f,0.f,0.f};

    int eb = base[e];
    int aRow = eb + m0 + wm + (lane & 15);
    int aK   = (lane >> 4) * 8;
    for (int k0 = 0; k0 < FDIM; k0 += 32) {
        FragU a[4], b[2];
#pragma unroll
        for (int i = 0; i < 4; i++) {
            const unsigned short* pa = Hpool + (size_t)(aRow + i * 16) * FDIM + k0 + aK;
            a[i].q[0] = *(const uint4*)pa; a[i].q[1] = *(const uint4*)(pa + 16);
        }
#pragma unroll
        for (int j = 0; j < 2; j++) {
            const unsigned short* pb = Bw + (size_t)(k0 + aK) * CDIM + n0 + wn + j * 16 + (lane & 15);
            for (int s = 0; s < 16; s++) ((unsigned short*)&b[j])[s] = pb[(size_t)((s & 7) + (s >> 3) * 16) * CDIM];
        }
#pragma unroll
        for (int i = 0; i < 4; i++)
#pragma unroll
            for (int j = 0; j < 2; j++)
                acc[i][j] = __builtin_amdgcn_wmma_f32_16x16x32_bf16(
                    false, a[i].v, false, b[j].v, (short)0, acc[i][j], false, false);
    }
    int cs = lane & 15, rs = (lane >> 4) * 8;
#pragma unroll
    for (int i = 0; i < 4; i++)
#pragma unroll
        for (int j = 0; j < 2; j++) {
            int col = n0 + wn + j * 16 + cs;
            float bias = b2[e * CDIM + col];
#pragma unroll
            for (int v = 0; v < 8; v++) {
                int row = m0 + wm + i * 16 + rs + v;
                if (row < M)
                    Ypool[(size_t)(eb + row) * CDIM + col] = acc[i][j][v] + bias;
            }
        }
}

// ---------------- combine: res += g0*Y[e0][p0] + g1*Y[e1][p1] ---------------
__global__ __launch_bounds__(256) void k_combine(const float* __restrict__ Ypool,
                                                 const int* __restrict__ slot_e,
                                                 const int* __restrict__ slot_pos,
                                                 const float* __restrict__ slot_g,
                                                 const int* __restrict__ base,
                                                 float* __restrict__ res) {
    int i = blockIdx.x * 256 + threadIdx.x;    // over NTOK*CDIM
    int t = i >> 10, c = i & 1023;
    int e0 = slot_e[2 * t], e1 = slot_e[2 * t + 1];
    float g0 = slot_g[2 * t], g1 = slot_g[2 * t + 1];
    int p0 = slot_pos[2 * t], p1 = slot_pos[2 * t + 1];
    float add = g0 * Ypool[(size_t)(base[e0] + p0) * CDIM + c]
              + g1 * Ypool[(size_t)(base[e1] + p1) * CDIM + c];
    res[i] += add;
}

extern "C" void kernel_launch(void* const* d_in, const int* in_sizes, int n_in,
                              void* d_out, int out_size, void* d_ws, size_t ws_size,
                              hipStream_t stream) {
    const float* x  = (const float*)d_in[0];
    const float* Ws = (const float*)d_in[1];
    const float* bs = (const float*)d_in[2];
    const float* Wg = (const float*)d_in[3];
    const float* bg = (const float*)d_in[4];
    const float* W1 = (const float*)d_in[5];
    const float* b1 = (const float*)d_in[6];
    const float* W2 = (const float*)d_in[7];
    const float* b2 = (const float*)d_in[8];
    // d_in[9] is k == 2 (compile-time assumed)
    float* res = (float*)d_out;
    float* out_fp = res + (size_t)NTOK * CDIM;

    // Workspace carve-out (~248 MB): bf16 copies, expert pools, routing arrays
    char* ws = (char*)d_ws;
    size_t off = 0;
    auto carve = [&](size_t bytes) { size_t o = off; off = (off + bytes + 255) & ~(size_t)255; return o; };
    unsigned short* Xb    = (unsigned short*)(ws + carve((size_t)NTOK * CDIM * 2));
    unsigned short* WsSum = (unsigned short*)(ws + carve((size_t)CDIM * CDIM * 2));
    float*          bsSum = (float*)         (ws + carve((size_t)CDIM * 4));
    unsigned short* W1b   = (unsigned short*)(ws + carve((size_t)EDIM * CDIM * FDIM * 2));
    unsigned short* W2b   = (unsigned short*)(ws + carve((size_t)EDIM * FDIM * CDIM * 2));
    unsigned short* Hpool = (unsigned short*)(ws + carve((size_t)POOLR * FDIM * 2));
    float*          Ypool = (float*)         (ws + carve((size_t)POOLR * CDIM * 4));
    int*   slot_e    = (int*)  (ws + carve(NSLOT * 4));
    int*   slot_pos  = (int*)  (ws + carve(NSLOT * 4));
    float* slot_g    = (float*)(ws + carve(NSLOT * 4));
    float* slot_ssel = (float*)(ws + carve(NSLOT * 4));
    int*   row_token = (int*)  (ws + carve(EDIM * MAXR * 4));
    int*   cnt       = (int*)  (ws + carve(EDIM * 4));
    int*   base      = (int*)  (ws + carve(EDIM * 4));
    int*   cntB      = (int*)  (ws + carve(BQ * EDIM * 4));
    float* psumB     = (float*)(ws + carve(BQ * EDIM * 4));
    (void)in_sizes; (void)n_in; (void)out_size; (void)ws_size;

    // 1) fp32 -> bf16 conversions (x, W1, W2) and shared-slab fold
    {
        int n = NTOK * CDIM;
        k_cvt_bf16<<<n / (256 * 4), 256, 0, stream>>>(x, Xb, n);
    }
    {
        int n = EDIM * CDIM * FDIM;
        k_cvt_bf16<<<n / (256 * 4), 256, 0, stream>>>(W1, W1b, n);
        k_cvt_bf16<<<n / (256 * 4), 256, 0, stream>>>(W2, W2b, n);
    }
    k_prep_ws<<<(CDIM * CDIM) / 256, 256, 0, stream>>>(Ws, bs, WsSum, bsSum);

    // 2) gate + deterministic routing + stats
    k_gate<<<NTOK / 8, 256, 0, stream>>>(x, Wg, bg, slot_e, slot_g, slot_ssel);
    k_assign<<<EDIM, 256, 0, stream>>>(slot_e, slot_ssel, slot_pos, row_token, cnt, cntB, psumB);
    k_finalize<<<1, 32, 0, stream>>>(cnt, cntB, psumB, base, out_fp);

    // 3) shared path GEMM (writes res = x + shared)
    k_gemm_shared<<<dim3(NTOK / 128, CDIM / 128), 256, 0, stream>>>(Xb, WsSum, bsSum, x, res);

    // 4) routed expert FFN (worst-case grids; blocks past cnt[e] exit uniformly)
    k_ffn1<<<dim3(MAXR / 128, FDIM / 128, EDIM), 256, 0, stream>>>(Xb, W1b, b1, row_token, cnt, base, Hpool);
    k_ffn2<<<dim3(MAXR / 128, CDIM / 128, EDIM), 256, 0, stream>>>(Hpool, W2b, b2, cnt, base, Ypool);

    // 5) gather-based combine (deterministic, no float atomics)
    k_combine<<<(NTOK * CDIM) / 256, 256, 0, stream>>>(Ypool, slot_e, slot_pos, slot_g, base, res);
}
